// BCMEmulator_30090540876111
// MI455X (gfx1250) — matrix operand order repaired and checked by reading, not compiled, not run on hardware
//
#include <hip/hip_runtime.h>

// ---------------------------------------------------------------------------
// Fused TCN emulator for MI455X (gfx1250, wave32, WMMA).
//
// Strategy: receptive field of the whole network is 64 samples, so each
// workgroup computes a 128-wide time tile (plus 64-halo) through ALL layers
// in LDS, using v_wmma_f32_16x16x32_f16 for every conv (A = weight 16x32 f16
// tile, B = activation 32x16 f16 tile gathered at per-tap time offsets,
// C/D = f32 accum). Weights are pre-transposed once into d_ws as f16 [m][kk]
// (kk = tap*CH + c, zero-padded to multiples of 32) by a prep kernel.
// ---------------------------------------------------------------------------

typedef __attribute__((ext_vector_type(16))) _Float16 v16h;
typedef __attribute__((ext_vector_type(8)))  _Float16 v8h;
typedef __attribute__((ext_vector_type(8)))  float    v8f;

#define Bsz   128
#define Tsz   1024
#define CIN   15
#define EMB   8
#define CH    64
#define TT    128          // output times per workgroup
#define HALO  64           // total receptive field of backbone
#define WWIN  (TT + HALO)  // 192 time samples held in LDS
#define XS    40           // LDS row stride (halves) for 32-ch input buffer
#define FS    72           // LDS row stride (halves) for 64-ch buffers
#define NT16  (WWIN / 16)  // 12 N-tiles of 16 columns

// d_ws layout (halves): transposed f16 weights
#define OFF_W0A 0          // [64][96]   (kk = tap*32 + c, c<23 real)
#define OFF_W0B 6144       // [64][192]
#define OFF_W0R 18432      // [64][32]
#define OFF_WA  20480      // 4 x [64][192]
#define OFF_WB  69632      // 4 x [64][192]
#define WT_TOTAL 118784

#define SMEM_BYTES (WWIN*XS*2 + 2*WWIN*FS*2)   // sX + sF + sH = 70656

// ---------------------------------------------------------------------------
// Weight pre-transpose: f32 (O,I,K) -> f16 [m][kk], kk = tap*CHpad + c
// ---------------------------------------------------------------------------
extern "C" __global__ void tcn_prep(const float* __restrict__ w0a,
                                    const float* __restrict__ w0b,
                                    const float* __restrict__ w0r,
                                    const float* __restrict__ wa,
                                    const float* __restrict__ wb,
                                    _Float16* __restrict__ wt) {
  int i = blockIdx.x * 256 + threadIdx.x;
  if (i >= WT_TOTAL) return;
  float v = 0.f;
  if (i < OFF_W0B) {                       // w0a: 64 x 96
    int m = i / 96, kk = i - m * 96, tap = kk >> 5, c = kk & 31;
    if (c < 23) v = w0a[(m * 23 + c) * 3 + tap];
  } else if (i < OFF_W0R) {                // w0b: 64 x 192
    int j = i - OFF_W0B;
    int m = j / 192, kk = j - m * 192, tap = kk >> 6, c = kk & 63;
    v = w0b[(m * 64 + c) * 3 + tap];
  } else if (i < OFF_WA) {                 // w0r: 64 x 32 (1x1)
    int j = i - OFF_W0R;
    int m = j >> 5, c = j & 31;
    if (c < 23) v = w0r[m * 23 + c];
  } else if (i < OFF_WB) {                 // wa: 4 x 64 x 192
    int j = i - OFF_WA;
    int blk = j / 12288, jj = j - blk * 12288;
    int m = jj / 192, kk = jj - m * 192, tap = kk >> 6, c = kk & 63;
    v = wa[((blk * 64 + m) * 64 + c) * 3 + tap];
  } else {                                 // wb: 4 x 64 x 192
    int j = i - OFF_WB;
    int blk = j / 12288, jj = j - blk * 12288;
    int m = jj / 192, kk = jj - m * 192, tap = kk >> 6, c = kk & 63;
    v = wb[((blk * 64 + m) * 64 + c) * 3 + tap];
  }
  wt[i] = (_Float16)v;
}

// ---------------------------------------------------------------------------
// A-fragment (16x32 f16): lane L holds row M=L&15;
// half-wave g=L>>4 supplies K = g*16+{0..15} (elems 0..15).
// Weight rows are contiguous [m][kkd] halves in global memory.
// ---------------------------------------------------------------------------
static __device__ __forceinline__ v16h load_afrag(const _Float16* __restrict__ wrow,
                                                  int g, int sbase) {
  const _Float16* p = wrow + sbase + g * 16;
  v16h a = *(const v16h*)(p);
  return a;
}

// ---------------------------------------------------------------------------
// One WMMA conv pass over the 192-sample window.
//   S      : K-steps of 32 (3 for conv0a, 6 for 64-ch K=3 convs)
//   CHDIM  : contraction channels per tap (32 padded input / 64)
//   ASRC   : LDS row stride of source buffer
//   EXTRA  : fused 1x1 projection (conv0r) accumulated after ReLU
// ---------------------------------------------------------------------------
template <int S, int CHDIM, int ASRC, bool EXTRA>
static __device__ __forceinline__ void conv_pass(
    const _Float16* __restrict__ src, _Float16* __restrict__ dst,
    const _Float16* __restrict__ wg, int kkd, const float* __restrict__ bias,
    int d, bool do_relu, const _Float16* __restrict__ resid,
    const _Float16* __restrict__ wg2, const _Float16* __restrict__ src2,
    const float* __restrict__ bias2, int tid) {
  const int lane = tid & 31, wv = tid >> 5;
  const int g = lane >> 4, nl = lane & 15;
  const int mo = (wv & 3) * 16;   // 16-row block of output channels
  const int nst = wv >> 2;        // 0/1 -> even/odd N-tiles

  // Preload this wave's A fragments (shared across all 6 N-tiles).
  v16h af[S];
  const _Float16* wrow = wg + (mo + nl) * kkd;
#pragma unroll
  for (int s = 0; s < S; ++s) af[s] = load_afrag(wrow, g, s * 32);

  v16h af2;
  if (EXTRA) af2 = load_afrag(wg2 + (mo + nl) * 32, g, 0);

  float br[8], br2[8];
#pragma unroll
  for (int r = 0; r < 8; ++r) {
    br[r] = bias[mo + g * 8 + r];
    if (EXTRA) br2[r] = bias2[mo + g * 8 + r];
  }

#pragma unroll
  for (int j = 0; j < 6; ++j) {
    const int nt = nst + 2 * j;
    const int tcol = nt * 16 + nl;
    v8f acc;
#pragma unroll
    for (int r = 0; r < 8; ++r) acc[r] = br[r];

#pragma unroll
    for (int s = 0; s < S; ++s) {
      const int base = s * 32 + g * 16;
      const int tap = base / CHDIM;   // constant per (s,g)
      const int c0 = base % CHDIM;
      int lt = tcol - (2 - tap) * d;  // causal tap offset
      if (lt < 0) lt = 0;             // halo clamp (discarded region only)
      v16h bf = *(const v16h*)(src + lt * ASRC + c0);
      acc = __builtin_amdgcn_wmma_f32_16x16x32_f16(false, af[s], false, bf,
                                                   (short)0, acc, false, false);
    }
    if (do_relu) {
#pragma unroll
      for (int r = 0; r < 8; ++r) acc[r] = fmaxf(acc[r], 0.f);
    }
    if (EXTRA) {  // f = relu(conv0b) + (conv0r(xin) + b0r)
      v16h bf2 = *(const v16h*)(src2 + tcol * XS + g * 16);
      acc = __builtin_amdgcn_wmma_f32_16x16x32_f16(false, af2, false, bf2,
                                                   (short)0, acc, false, false);
#pragma unroll
      for (int r = 0; r < 8; ++r) acc[r] += br2[r];
    }
    if (resid) {
      v8h rf = *(const v8h*)(resid + tcol * FS + mo + g * 8);
#pragma unroll
      for (int r = 0; r < 8; ++r) acc[r] += (float)rf[r];
    }
    v8h o;
#pragma unroll
    for (int r = 0; r < 8; ++r) o[r] = (_Float16)acc[r];
    *(v8h*)(dst + tcol * FS + mo + g * 8) = o;
  }
}

static __device__ __forceinline__ float softplusf(float x) {
  return fmaxf(x, 0.f) + log1pf(expf(-fabsf(x)));
}

// ---------------------------------------------------------------------------
// Main fused kernel: grid = B * (T/TT) = 1024 blocks x 256 threads.
// ---------------------------------------------------------------------------
extern "C" __global__ void __launch_bounds__(256) tcn_main(
    const float* __restrict__ x, const int* __restrict__ fids,
    const float* __restrict__ femb, const float* __restrict__ b0a,
    const float* __restrict__ b0b, const float* __restrict__ b0r,
    const float* __restrict__ ba, const float* __restrict__ bb,
    const float* __restrict__ petw, const float* __restrict__ petb,
    const float* __restrict__ pckw, const float* __restrict__ pckb,
    const float* __restrict__ aetw, const float* __restrict__ aetb,
    const _Float16* __restrict__ wt, float* __restrict__ out) {
  extern __shared__ char smem[];
  _Float16* sX = (_Float16*)smem;                                 // [192][40]
  _Float16* sF = (_Float16*)(smem + WWIN * XS * 2);               // [192][72]
  _Float16* sH = (_Float16*)(smem + WWIN * XS * 2 + WWIN * FS * 2);

  const int tid = threadIdx.x;
  const int b = blockIdx.x >> 3;
  const int t0 = (blockIdx.x & 7) * TT;
  const int fid = fids[b];

  // Stage padded input xin (23 real ch -> 32) for window [t0-64, t0+128).
  for (int u = tid; u < WWIN * 32; u += 256) {
    const int c = u / WWIN, lt = u - c * WWIN;
    float v;
    if (c < CIN) {
      const int gt = t0 - HALO + lt;
      v = (gt >= 0) ? x[(b * CIN + c) * Tsz + gt] : 0.f;
    } else if (c < CIN + EMB) {
      v = femb[fid * EMB + (c - CIN)];
    } else {
      v = 0.f;
    }
    sX[lt * XS + c] = (_Float16)v;
  }
  __syncthreads();

  // Block 0: h = relu(conv0a(xin)), f = relu(conv0b(h)) + conv0r(xin)
  conv_pass<3, 32, XS, false>(sX, sH, wt + OFF_W0A, 96, b0a, 1, true, nullptr,
                              nullptr, nullptr, nullptr, tid);
  __syncthreads();
  conv_pass<6, 64, FS, true>(sH, sF, wt + OFF_W0B, 192, b0b, 1, true, nullptr,
                             wt + OFF_W0R, sX, b0r, tid);
  __syncthreads();

  // Residual blocks, dilation 2,4,8,16
  int d = 2;
  for (int i = 0; i < 4; ++i) {
    conv_pass<6, 64, FS, false>(sF, sH, wt + OFF_WA + i * 12288, 192,
                                ba + i * 64, d, true, nullptr, nullptr,
                                nullptr, nullptr, tid);
    __syncthreads();
    conv_pass<6, 64, FS, false>(sH, sF, wt + OFF_WB + i * 12288, 192,
                                bb + i * 64, d, true, sF, nullptr, nullptr,
                                nullptr, tid);
    __syncthreads();
    d <<= 1;
  }

  // Heads: stash head weights in LDS (reuse sX region, no longer needed).
  float* hw = (float*)sX;  // [0..63] pet_w, [64..127] pck_w, [128..193] aet_w
  if (tid < 64) { hw[tid] = petw[tid]; hw[64 + tid] = pckw[tid]; }
  if (tid < 66) hw[128 + tid] = aetw[tid];
  if (tid == 0) { hw[194] = petb[0]; hw[195] = pckb[0]; hw[196] = aetb[0]; }
  __syncthreads();

  if (tid < TT) {
    const int lt = HALO + tid;
    const int gt = t0 + tid;
    const _Float16* fr = sF + lt * FS;
    float sp = 0.f, sk = 0.f, sa = 0.f;
#pragma unroll
    for (int cb = 0; cb < 8; ++cb) {
      v8h v = *(const v8h*)(fr + cb * 8);
#pragma unroll
      for (int r = 0; r < 8; ++r) {
        const float fc = (float)v[r];
        const int c = cb * 8 + r;
        sp += hw[c] * fc;
        sk += hw[64 + c] * fc;
        sa += hw[128 + c] * fc;
      }
    }
    const float pet = softplusf(sp + hw[194]);
    const float pck = softplusf(sk + hw[195]);
    const float alin = sa + hw[128 + 64] * pet + hw[128 + 65] * pck + hw[196];
    const float aet = pet / (1.f + expf(-alin));
    const float cwd = pet - aet;
    const int BT = Bsz * Tsz;
    out[b * Tsz + gt] = pet;
    out[BT + b * Tsz + gt] = pck;
    out[2 * BT + b * Tsz + gt] = aet;
    out[3 * BT + b * Tsz + gt] = cwd;
  }
}

// ---------------------------------------------------------------------------
extern "C" void kernel_launch(void* const* d_in, const int* in_sizes, int n_in,
                              void* d_out, int out_size, void* d_ws,
                              size_t ws_size, hipStream_t stream) {
  (void)in_sizes; (void)n_in; (void)out_size; (void)ws_size;
  const float* x    = (const float*)d_in[0];
  const int*   fids = (const int*)d_in[1];
  const float* femb = (const float*)d_in[2];
  const float* w0a  = (const float*)d_in[3];
  const float* b0a  = (const float*)d_in[4];
  const float* w0b  = (const float*)d_in[5];
  const float* b0b  = (const float*)d_in[6];
  const float* w0r  = (const float*)d_in[7];
  const float* b0r  = (const float*)d_in[8];
  const float* wa   = (const float*)d_in[9];
  const float* ba   = (const float*)d_in[10];
  const float* wb   = (const float*)d_in[11];
  const float* bb   = (const float*)d_in[12];
  const float* petw = (const float*)d_in[13];
  const float* petb = (const float*)d_in[14];
  const float* pckw = (const float*)d_in[15];
  const float* pckb = (const float*)d_in[16];
  const float* aetw = (const float*)d_in[17];
  const float* aetb = (const float*)d_in[18];
  _Float16* wt = (_Float16*)d_ws;   // 237,568 B of transposed f16 weights
  float* out = (float*)d_out;

  tcn_prep<<<(WT_TOTAL + 255) / 256, 256, 0, stream>>>(w0a, w0b, w0r, wa, wb, wt);

  (void)hipFuncSetAttribute(reinterpret_cast<const void*>(tcn_main),
                            hipFuncAttributeMaxDynamicSharedMemorySize,
                            SMEM_BYTES);
  tcn_main<<<Bsz * (Tsz / TT), 256, SMEM_BYTES, stream>>>(
      x, fids, femb, b0a, b0b, b0r, ba, bb, petw, petb, pckw, pckb, aetw, aetb,
      wt, out);
}
